// bilstm_crf_38268158608004
// MI455X (gfx1250) — matrix operand order repaired and checked by reading, not compiled, not run on hardware
//
#include <hip/hip_runtime.h>
#include <hip/hip_bf16.h>

typedef __attribute__((ext_vector_type(8)))  __bf16 v8bf;
typedef __attribute__((ext_vector_type(16))) __bf16 v16bf;
typedef __attribute__((ext_vector_type(8)))  float  v8f;

typedef unsigned int v4u __attribute__((ext_vector_type(4)));
typedef int          v4i __attribute__((ext_vector_type(4)));
typedef int          v8i __attribute__((ext_vector_type(8)));

#if defined(__has_builtin)
#if __has_builtin(__builtin_amdgcn_tensor_load_to_lds) && \
    __has_builtin(__builtin_amdgcn_s_wait_tensorcnt)
#define HAVE_TDM 1
#else
#define HAVE_TDM 0
#endif
#else
#define HAVE_TDM 0
#endif

constexpr int kE  = 1024;
constexpr int kH  = 1024;
constexpr int kT  = 8192;
constexpr int kK  = 5;
constexpr int kFH = 4 * kH;            // 4096 gate rows
constexpr int kStop = 4;               // START=3, STOP=4
constexpr int NWG_DIR = 128;           // recurrence WGs per direction
constexpr int UPW = kH / NWG_DIR;      // 8 hidden units per WG (32 Whh rows)

// ---------------------------------------------------------------- fp32 -> bf16 (RNE)
__global__ void f2bf_kernel(const float* __restrict__ src,
                            unsigned short* __restrict__ dst, int n) {
  int i = blockIdx.x * blockDim.x + threadIdx.x;
  if (i < n) {
    unsigned u = __float_as_uint(src[i]);
    unsigned r = u + 0x7FFFu + ((u >> 16) & 1u);
    dst[i] = (unsigned short)(r >> 16);
  }
}

// ---------------------------------------------------------------- init (counters + h state)
__global__ void init_kernel(const float* __restrict__ h0,
                            float* __restrict__ hcur, unsigned int* cnt) {
  int i = blockIdx.x * blockDim.x + threadIdx.x;
  if (i < 2 * kH) hcur[i] = h0[i];
  if (i < 8)      cnt[i] = 0u;
}

// ---------------------------------------------------------------- xg = x @ Wih^T + (bih+bhh)
// One wave computes a 16(M) x 64(N) strip; 8 waves per block; blockIdx.z = direction.
__global__ __launch_bounds__(256) void xg_gemm_kernel(
    const unsigned short* __restrict__ xbf,
    const unsigned short* __restrict__ wfbf,
    const unsigned short* __restrict__ wbbf,
    const float* __restrict__ bihf, const float* __restrict__ bhhf,
    const float* __restrict__ bihb, const float* __restrict__ bhhb,
    float* __restrict__ xgf, float* __restrict__ xgb) {
  const int dir = blockIdx.z;
  const unsigned short* wbf = dir ? wbbf : wfbf;
  const float* bih = dir ? bihb : bihf;
  const float* bhh = dir ? bhhb : bhhf;
  float* xg = dir ? xgb : xgf;

  const int wave = threadIdx.x >> 5;
  const int lane = threadIdx.x & 31;
  const int task = blockIdx.x * 8 + wave;   // 512 mtiles * 64 ngroups = 32768 tasks
  const int mtile = task & 511;
  const int ngrp  = task >> 9;
  const int m0 = mtile * 16;
  const int n0 = ngrp * 64;
  const int lm = lane & 15;
  const int half = lane >> 4;

  const unsigned short* arow = xbf + (size_t)(m0 + lm) * kE;
  v8f acc[4] = {v8f{}, v8f{}, v8f{}, v8f{}};

  for (int kb = 0; kb < kE; kb += 32) {
    // A 16x32 bf16 fragment: two 16B runs per lane (K interleave per ISA layout)
    v8bf a_lo = *(const v8bf*)(arow + kb + half * 8);
    v8bf a_hi = *(const v8bf*)(arow + kb + 16 + half * 8);
    v16bf a = __builtin_shufflevector(a_lo, a_hi,
                0, 1, 2, 3, 4, 5, 6, 7, 8, 9, 10, 11, 12, 13, 14, 15);
#pragma unroll
    for (int nt = 0; nt < 4; ++nt) {
      // B 32x16 bf16 fragment: two 16B runs per lane (K interleave matching A)
      const unsigned short* brow =
          wbf + (size_t)(n0 + nt * 16 + lm) * kE + kb + half * 8;
      v8bf b_lo = *(const v8bf*)(brow);
      v8bf b_hi = *(const v8bf*)(brow + 16);
      v16bf b = __builtin_shufflevector(b_lo, b_hi,
                  0, 1, 2, 3, 4, 5, 6, 7, 8, 9, 10, 11, 12, 13, 14, 15);
      acc[nt] = __builtin_amdgcn_wmma_f32_16x16x32_bf16(
          false, a, false, b, (short)0, acc[nt], false, false);
    }
  }

#pragma unroll
  for (int nt = 0; nt < 4; ++nt) {
    int col = n0 + nt * 16 + lm;
    float bias = bih[col] + bhh[col];
#pragma unroll
    for (int r = 0; r < 8; ++r) {
      int m = m0 + half * 8 + r;                 // C/D layout: lanes 0-15 -> M 0..7, 16-31 -> M 8..15
      xg[(size_t)m * kFH + col] = acc[nt][r] + bias;
    }
  }
}

// ---------------------------------------------------------------- persistent LSTM recurrence
__global__ __launch_bounds__(256) void lstm_rec_kernel(
    const float* __restrict__ xgf, const float* __restrict__ xgb,
    const unsigned short* __restrict__ whhf, const unsigned short* __restrict__ whhb,
    const float* __restrict__ c0,
    float* __restrict__ hf, float* __restrict__ hb,
    float* hcur, unsigned int* cnt) {
  __shared__ float hlds[kH];
  __shared__ float glds[4 * UPW];

  const int tid = threadIdx.x;
  const int wg  = blockIdx.x;                 // 0..2*NWG_DIR-1
  const int dir = (wg >= NWG_DIR) ? 1 : 0;
  const int wgd = dir ? wg - NWG_DIR : wg;
  const int u0  = wgd * UPW;
  const float* xg = dir ? xgb : xgf;
  const unsigned short* whh = dir ? whhb : whhf;
  float* hsout = dir ? hb : hf;
  float* hslot = hcur + dir * kH;
  unsigned int* mycnt = cnt + dir;

  const int wave = tid >> 5, lane = tid & 31;

  float c = 0.f;
  if (tid < UPW) c = c0[dir * kH + u0 + tid];

#if HAVE_TDM
  // TDM descriptor: 1-row copy of 512 x 8B (= h_prev, 4KB) from global hslot -> LDS hlds.
  typedef __attribute__((address_space(3))) float lds_float;
  unsigned lds_base = (unsigned)(unsigned long long)(lds_float*)hlds;
  unsigned long long ga = (unsigned long long)hslot;
  v4u g0;
  g0.x = 1u;                                       // count=1 (valid user D#)
  g0.y = lds_base;                                 // lds_addr (bytes)
  g0.z = (unsigned)(ga & 0xffffffffull);           // global_addr[31:0]
  g0.w = (unsigned)((ga >> 32) & 0x01ffffffull)    // global_addr[56:32]
         | (2u << 30);                             // type=2 ("image")
  v8i g1;
  g1[0] = (int)(3u << 16);                         // wg_mask=0, data_size=3 (8B)
  g1[1] = (int)(512u << 16);                       // tensor_dim0[15:0]=512
  g1[2] = (int)(1u << 16);                         // tensor_dim0[31:16]=0 | tensor_dim1[15:0]=1
  g1[3] = (int)(512u << 16);                       // tensor_dim1[31:16]=0 | tile_dim0=512
  g1[4] = 1;                                       // tile_dim1=1, tile_dim2=0
  g1[5] = 512;                                     // tensor_dim0_stride[31:0]
  g1[6] = (int)(512u << 16);                       // stride0 hi=0 | tensor_dim1_stride[15:0]
  g1[7] = 0;
  v4i gz4 = {0, 0, 0, 0};
  v8i gz8 = {0, 0, 0, 0, 0, 0, 0, 0};
#endif

  for (int s = 0; s < kT; ++s) {
#if HAVE_TDM
    // stage h_prev into LDS via Tensor Data Mover (wave 0 only; TENSORcnt-tracked)
    if (tid < 32) {
      __builtin_amdgcn_tensor_load_to_lds(g0, g1, gz4, gz4, gz8, 0);
      __builtin_amdgcn_s_wait_tensorcnt(0);
    }
#else
    for (int i = tid; i < kH; i += 256)
      hlds[i] = __hip_atomic_load(&hslot[i], __ATOMIC_RELAXED, __HIP_MEMORY_SCOPE_AGENT);
#endif
    __syncthreads();

    const int trow = dir ? (kT - 1 - s) : s;

    // 32 gate rows for this WG: r = u*4 + g, gate order i,f,g,o
#pragma unroll
    for (int j = 0; j < 4; ++j) {
      const int r = wave * 4 + j;
      const int u = r >> 2, g = r & 3;
      const unsigned int* wrow =
          (const unsigned int*)(whh + ((size_t)(g * kH + u0 + u)) * kH);
      float acc = 0.f;
#pragma unroll
      for (int it = 0; it < 16; ++it) {
        const int idx = it * 32 + lane;        // 512 bf16-pairs per row
        unsigned int w2 = wrow[idx];
        float lo = __uint_as_float(w2 << 16);
        float hi = __uint_as_float(w2 & 0xFFFF0000u);
        acc += lo * hlds[2 * idx] + hi * hlds[2 * idx + 1];
      }
      for (int off = 16; off; off >>= 1) acc += __shfl_xor(acc, off, 32);
      if (lane == 0)
        glds[r] = acc + xg[(size_t)trow * kFH + (size_t)g * kH + u0 + u];
    }
    __syncthreads();

    if (tid < UPW) {
      float gi = glds[tid * 4 + 0], gf = glds[tid * 4 + 1];
      float gg = glds[tid * 4 + 2], go = glds[tid * 4 + 3];
      float i_ = 1.f / (1.f + __expf(-gi));
      float f_ = 1.f / (1.f + __expf(-gf));
      float g_ = tanhf(gg);
      float o_ = 1.f / (1.f + __expf(-go));
      c = f_ * c + i_ * g_;
      float h = o_ * tanhf(c);
      hsout[(size_t)trow * kH + u0 + tid] = h;
      __hip_atomic_store(&hslot[u0 + tid], h, __ATOMIC_RELEASE, __HIP_MEMORY_SCOPE_AGENT);
    }
    __syncthreads();

    // per-direction grid barrier (monotonic counter; zeroed by init_kernel each launch)
    if (tid == 0) {
      __hip_atomic_fetch_add(mycnt, 1u, __ATOMIC_ACQ_REL, __HIP_MEMORY_SCOPE_AGENT);
      const unsigned int target = (unsigned int)(s + 1) * NWG_DIR;
      while (__hip_atomic_load(mycnt, __ATOMIC_ACQUIRE, __HIP_MEMORY_SCOPE_AGENT) < target)
        __builtin_amdgcn_s_sleep(1);
    }
    __syncthreads();
  }
}

// ---------------------------------------------------------------- feats = [hf|hb] @ Wtag^T + b
__global__ __launch_bounds__(160) void feats_kernel(
    const float* __restrict__ hf, const float* __restrict__ hb,
    const float* __restrict__ wtag, const float* __restrict__ btag,
    float* __restrict__ feats) {
  const int t = blockIdx.x;
  const int k = threadIdx.x >> 5, lane = threadIdx.x & 31;
  const float* wk  = wtag + (size_t)k * 2 * kH;
  const float* hft = hf + (size_t)t * kH;
  const float* hbt = hb + (size_t)t * kH;
  float acc = 0.f;
  for (int i = lane; i < kH; i += 32)
    acc += hft[i] * wk[i] + hbt[i] * wk[kH + i];
  for (int off = 16; off; off >>= 1) acc += __shfl_xor(acc, off, 32);
  if (lane == 0) feats[(size_t)t * kK + k] = acc + btag[k];
}

// ---------------------------------------------------------------- Viterbi (single wave)
__global__ __launch_bounds__(32) void viterbi_kernel(
    const float* __restrict__ feats, const float* __restrict__ trans,
    float* __restrict__ out) {
  __shared__ unsigned int bpw[kT];            // 4-bit backpointer per tag, packed
  const int lane = threadIdx.x;
  float tr[kK];
  for (int p = 0; p < kK; ++p) tr[p] = 0.f;
  if (lane < kK)
    for (int p = 0; p < kK; ++p) tr[p] = trans[lane * kK + p];

  float fv = (lane == 3) ? 0.f : -1e8f;       // START = 3

  for (int t = 0; t < kT; ++t) {
    float best = -3.4e38f; int bp = 0;
#pragma unroll
    for (int p = 0; p < kK; ++p) {
      float fvp = __shfl(fv, p, 32);
      float sc = fvp + tr[p];
      if (sc > best) { best = sc; bp = p; }
    }
    unsigned int word = 0;
#pragma unroll
    for (int j = 0; j < kK; ++j)
      word |= ((unsigned)__shfl(bp, j, 32) & 0xFu) << (4 * j);
    if (lane == 0) bpw[t] = word;
    float feat = (lane < kK) ? feats[(size_t)t * kK + lane] : 0.f;
    fv = best + feat;
    if (lane >= kK) fv = -1e8f;
  }

  float term = (lane < kK) ? fv + trans[kStop * kK + lane] : -3.4e38f;
  float bestv = term; int besti = lane;
  for (int off = 16; off; off >>= 1) {
    float ov = __shfl_xor(bestv, off, 32);
    int   oi = __shfl_xor(besti, off, 32);
    if (ov > bestv || (ov == bestv && oi < besti)) { bestv = ov; besti = oi; }
  }
  __syncthreads();
  if (lane == 0) {
    out[0] = bestv;
    int tag = besti;
    out[1 + (kT - 1)] = (float)tag;
    for (int t = kT - 2; t >= 0; --t) {
      tag = (int)((bpw[t + 1] >> (4 * tag)) & 0xFu);
      out[1 + t] = (float)tag;
    }
  }
}

// ---------------------------------------------------------------- host
extern "C" void kernel_launch(void* const* d_in, const int* in_sizes, int n_in,
                              void* d_out, int out_size, void* d_ws, size_t ws_size,
                              hipStream_t stream) {
  const float* x     = (const float*)d_in[0];
  const float* wihf  = (const float*)d_in[1];
  const float* whhf  = (const float*)d_in[2];
  const float* bihf  = (const float*)d_in[3];
  const float* bhhf  = (const float*)d_in[4];
  const float* wihb  = (const float*)d_in[5];
  const float* whhb  = (const float*)d_in[6];
  const float* bihb  = (const float*)d_in[7];
  const float* bhhb  = (const float*)d_in[8];
  const float* wtag  = (const float*)d_in[9];
  const float* btag  = (const float*)d_in[10];
  const float* trans = (const float*)d_in[11];
  const float* h0    = (const float*)d_in[12];
  const float* c0    = (const float*)d_in[13];

  size_t off = 0;
  char* base = (char*)d_ws;
  auto carve = [&](size_t bytes) -> void* {
    void* p = base + off;
    off += (bytes + 255) & ~(size_t)255;
    return p;
  };
  unsigned short* xbf    = (unsigned short*)carve((size_t)kT * kE * 2);
  unsigned short* wihfbf = (unsigned short*)carve((size_t)kFH * kE * 2);
  unsigned short* wihbbf = (unsigned short*)carve((size_t)kFH * kE * 2);
  unsigned short* whhfbf = (unsigned short*)carve((size_t)kFH * kH * 2);
  unsigned short* whhbbf = (unsigned short*)carve((size_t)kFH * kH * 2);
  float* xgf   = (float*)carve((size_t)kT * kFH * 4);
  float* xgb   = (float*)carve((size_t)kT * kFH * 4);
  float* hf    = (float*)carve((size_t)kT * kH * 4);
  float* hb    = (float*)carve((size_t)kT * kH * 4);
  float* hcur  = (float*)carve((size_t)2 * kH * 4);
  float* feats = (float*)carve((size_t)kT * kK * 4);
  unsigned int* cnt = (unsigned int*)carve(256);

  init_kernel<<<8, 256, 0, stream>>>(h0, hcur, cnt);

  f2bf_kernel<<<(kT * kE + 255) / 256, 256, 0, stream>>>(x, xbf, kT * kE);
  f2bf_kernel<<<(kFH * kE + 255) / 256, 256, 0, stream>>>(wihf, wihfbf, kFH * kE);
  f2bf_kernel<<<(kFH * kE + 255) / 256, 256, 0, stream>>>(wihb, wihbbf, kFH * kE);
  f2bf_kernel<<<(kFH * kH + 255) / 256, 256, 0, stream>>>(whhf, whhfbf, kFH * kH);
  f2bf_kernel<<<(kFH * kH + 255) / 256, 256, 0, stream>>>(whhb, whhbbf, kFH * kH);

  xg_gemm_kernel<<<dim3(4096, 1, 2), 256, 0, stream>>>(
      xbf, wihfbf, wihbbf, bihf, bhhf, bihb, bhhb, xgf, xgb);

  lstm_rec_kernel<<<2 * NWG_DIR, 256, 0, stream>>>(
      xgf, xgb, whhfbf, whhbbf, c0, hf, hb, hcur, cnt);

  feats_kernel<<<kT, 160, 0, stream>>>(hf, hb, wtag, btag, feats);

  viterbi_kernel<<<1, 32, 0, stream>>>(feats, trans, (float*)d_out);
}
